// Heat2D_26637387169875
// MI455X (gfx1250) — matrix operand order repaired and checked by reading, not compiled, not run on hardware
//
#include <hip/hip_runtime.h>
#include <hip/hip_bf16.h>

#define RES   128
#define DIM   96
#define HID   96
#define BATCH 16
#define PIXC  (RES * HID)   // 12288 columns for the H-direction DCT GEMMs
#define PI_F  3.14159265358979323846f
#define LOG2E 1.4426950408889634f

typedef _Float16 f16;
typedef __attribute__((ext_vector_type(16))) _Float16 v16h;
typedef __attribute__((ext_vector_type(8)))  _Float16 v8h;
typedef __attribute__((ext_vector_type(8)))  float    v8f;

// ---------------- CDNA5 async global<->LDS (ASYNCcnt) ----------------
// 16B per lane per op; LDS address = low 32 bits of generic pointer

__device__ __forceinline__ uint32_t lds_off_u32(const void* p) {
    return (uint32_t)(uintptr_t)p;
}
__device__ __forceinline__ void async_ld16(void* lds, const void* gptr) {
    asm volatile("global_load_async_to_lds_b128 %0, %1, off"
                 :: "v"(lds_off_u32(lds)), "v"(gptr) : "memory");
}
__device__ __forceinline__ void async_st16(void* gptr, const void* lds) {
    asm volatile("global_store_async_from_lds_b128 %0, %1, off"
                 :: "v"(gptr), "v"(lds_off_u32(lds)) : "memory");
}
__device__ __forceinline__ void async_wait_all() {
    asm volatile("s_wait_asynccnt 0x0" ::: "memory");
}

// ---------------- WMMA helpers (wave32, 16x16x32 f16 -> f32) ----------------

__device__ __forceinline__ v8f vzero8() {
    v8f z;
#pragma unroll
    for (int i = 0; i < 8; ++i) z[i] = 0.0f;
    return z;
}

__device__ __forceinline__ v8f wmma16(v16h a, v16h b, v8f c) {
    return __builtin_amdgcn_wmma_f32_16x16x32_f16(
        /*neg_a=*/false, a, /*neg_b=*/false, b,
        /*c_mod=*/(short)0, c, /*reuse_a=*/false, /*reuse_b=*/false);
}

// A-frag from row-major [M x K] tile (LDS). ISA 7.12.2 16-bit A layout:
// lane L: row = L%16, half = L/16; two contiguous 8-halfword runs.
__device__ __forceinline__ v16h load_a_frag(const f16* A, int lda, int k0, int lane) {
    int row = lane & 15, half = lane >> 4;
    const f16* p = A + (size_t)row * lda + k0 + half * 8;
    union { v16h v; v8h h[2]; } u;
    u.h[0] = *(const v8h*)p;
    u.h[1] = *(const v8h*)(p + 16);
    return u.v;
}

// B-frag from TRANSPOSED operand stored [N x K] row-major (LDS): contiguous b128s.
// lane L: N = L%16, half = L/16; two contiguous 8-halfword runs (K = half*8.., 16+half*8..).
__device__ __forceinline__ v16h load_b_frag_nk(const f16* BT, int ldk, int nbase, int k0, int lane) {
    int n = lane & 15, half = lane >> 4;
    const f16* p = BT + (size_t)(nbase + n) * ldk + k0 + half * 8;
    union { v16h v; v8h h[2]; } u;
    u.h[0] = *(const v8h*)p;
    u.h[1] = *(const v8h*)(p + 16);
    return u.v;
}

// ---------------- K1: constants (cos maps, log2 decay, f16 weights) ----------------

__global__ void k_precompute(const float* __restrict__ lin_w, const float* __restrict__ out_w,
                             f16* __restrict__ cosn, f16* __restrict__ cosnT,
                             float* __restrict__ lwexp,
                             f16* __restrict__ linw_h, f16* __restrict__ outw_h) {
    int tid = blockIdx.x * blockDim.x + threadIdx.x;   // grid covers 18432
    if (tid < RES * RES) {
        int n = tid / RES, h = tid % RES;
        float scale = (n == 0) ? rsqrtf((float)RES) : rsqrtf((float)RES * 0.5f);
        float val = cosf(((float)h + 0.5f) * (float)n * (PI_F / (float)RES)) * scale;
        cosn[n * RES + h]  = (f16)val;
        cosnT[h * RES + n] = (f16)val;
        float an = (float)n / RES; an *= an;
        float ah = (float)h / RES; ah *= ah;
        float d  = sqrtf(an + ah);
        float lw = -10.0f * d * LOG2E;                 // log2(exp(-10 d))
        float lmin = log2f(1e-4f);
        lwexp[tid] = lw < lmin ? lmin : lw;
    }
    if (tid < 2 * HID * DIM) linw_h[tid] = (f16)lin_w[tid];
    if (tid < DIM * HID)     outw_h[tid] = (f16)out_w[tid];
}

// ---------------- K2: k = relu(freq_embed @ tok_w^T + tok_b)  [16 x 96] ----------------

__global__ void k_tok(const float* __restrict__ fe, const float* __restrict__ tok_w,
                      const float* __restrict__ tok_b, float* __restrict__ kk) {
    int tid = blockIdx.x * blockDim.x + threadIdx.x;
    if (tid >= BATCH * DIM) return;
    int b = tid / DIM, o = tid % DIM;
    float acc = tok_b[o];
#pragma unroll 4
    for (int c = 0; c < DIM; ++c) acc += fe[b * DIM + c] * tok_w[o * DIM + c];
    kk[tid] = acc > 0.0f ? acc : 0.0f;
}

// ---------------- K3: depthwise 3x3 conv, NCHW f32 -> [b,h,w,c] f16 ----------------

__global__ void k_dwconv(const float* __restrict__ x, const float* __restrict__ w9,
                         const float* __restrict__ bias, f16* __restrict__ y) {
    // grid = BATCH*RES (one (b,h) row per block), 256 threads
    __shared__ float wl[DIM * 9];
    __shared__ float bl[DIM];
    __shared__ __align__(16) f16 tile[RES][DIM];       // [w][c]
    int blk = blockIdx.x;
    int b = blk / RES, h = blk % RES;
    int t = threadIdx.x;
    for (int i = t; i < DIM * 9; i += 256) wl[i] = w9[i];
    for (int i = t; i < DIM;     i += 256) bl[i] = bias[i];
    __syncthreads();
    int wp = t & 127;
    int cg = t >> 7;                                   // 0..1
    const float* xb = x + (size_t)b * DIM * RES * RES;
#pragma unroll 2
    for (int i = 0; i < DIM / 2; ++i) {
        int c = cg * (DIM / 2) + i;
        const float* xc = xb + (size_t)c * RES * RES;
        float acc = bl[c];
#pragma unroll
        for (int kh = -1; kh <= 1; ++kh) {
            int hh = h + kh;
            if (hh < 0 || hh >= RES) continue;
#pragma unroll
            for (int kw = -1; kw <= 1; ++kw) {
                int ww = wp + kw;
                if (ww < 0 || ww >= RES) continue;
                acc += wl[c * 9 + (kh + 1) * 3 + (kw + 1)] * xc[hh * RES + ww];
            }
        }
        tile[wp][c] = (f16)acc;
    }
    __syncthreads();
    // async DMA out: [w][c] tile == target layout; 1536 x 16B chunks, 6/thread
    f16* yrow = y + (size_t)(b * RES + h) * RES * DIM;
    f16* tf = &tile[0][0];
#pragma unroll
    for (int j = 0; j < 6; ++j) {
        int l = t + j * 256;
        async_st16(yrow + l * 8, tf + l * 8);
    }
}

// ---------------- K4: xl = y @ lin_w^T + lin_b ; split -> xh, z (f16) ----------------

__global__ void k_lin(const f16* __restrict__ y, const f16* __restrict__ linw,
                      const float* __restrict__ lin_b,
                      f16* __restrict__ xh, f16* __restrict__ z) {
    // grid = (BATCH*RES*RES)/64; 256 threads = 8 waves; tile: 64 pixels x 192 outs
    __shared__ __align__(16) f16 At[64][DIM];          // 12 KB, [pixel][c] ; reused: xh stage
    __shared__ __align__(16) f16 Bt[2 * HID][DIM];     // 36 KB, [o][c]    ; reused: z stage
    __shared__ float biasl[2 * HID];
    int t = threadIdx.x;
    size_t pixbase = (size_t)blockIdx.x * 64;
    f16* Af = &At[0][0];
    f16* Bf = &Bt[0][0];
#pragma unroll
    for (int j = 0; j < 3; ++j) {                      // 768 chunks of activations
        int l = t + j * 256;
        async_ld16(Af + l * 8, y + pixbase * DIM + l * 8);
    }
#pragma unroll
    for (int j = 0; j < 9; ++j) {                      // 2304 chunks of weights
        int l = t + j * 256;
        async_ld16(Bf + l * 8, linw + l * 8);
    }
    if (t < 2 * HID) biasl[t] = lin_b[t];
    async_wait_all();
    __syncthreads();
    int lane = t & 31, wv = t >> 5;
    int msub = wv & 3;                                 // 16-pixel sub-tile
    int nh   = wv >> 2;                                // 0..1 : six 16-wide n-tiles each
    v8f acc[6];
#pragma unroll
    for (int s = 0; s < 6; ++s) acc[s] = vzero8();
#pragma unroll
    for (int k0 = 0; k0 < DIM; k0 += 32) {
        v16h a = load_a_frag(&At[msub * 16][0], DIM, k0, lane);
#pragma unroll
        for (int s = 0; s < 6; ++s) {
            v16h bf = load_b_frag_nk(Bf, DIM, (nh * 6 + s) * 16, k0, lane);
            acc[s] = wmma16(a, bf, acc[s]);
        }
    }
    __syncthreads();                                   // frag reads done -> reuse LDS
    int col = lane & 15, half = lane >> 4;
#pragma unroll
    for (int s = 0; s < 6; ++s) {
        int o = (nh * 6 + s) * 16 + col;
        float bo = biasl[o];
#pragma unroll
        for (int r = 0; r < 8; ++r) {
            int prow = msub * 16 + r + 8 * half;
            float val = acc[s][r] + bo;
            if (o < HID) Af[prow * HID + o]         = (f16)val;
            else         Bf[prow * HID + (o - HID)] = (f16)val;
        }
    }
    __syncthreads();
#pragma unroll
    for (int j = 0; j < 3; ++j) {                      // 768 chunks each, contiguous
        int l = t + j * 256;
        async_st16(xh + pixbase * HID + l * 8, Af + l * 8);
        async_st16(z  + pixbase * HID + l * 8, Bf + l * 8);
    }
}

// ---------------- K5/K7: Y[b][n][p] = sum_h cosA[n][h] * X[b][h][p] ----------------

__global__ void k_dct_h(const f16* __restrict__ cosA, const f16* __restrict__ X,
                        f16* __restrict__ Y) {
    // grid = BATCH * (PIXC/64); 256 threads = 8 waves; block tile: 128 n x 64 p
    __shared__ __align__(16) f16 cosl[RES * RES];      // 32 KB; reused as out stage [128][64]
    __shared__ __align__(16) f16 xtT[64][32];          // 4 KB, B^T slab [p][k]
    int t = threadIdx.x;
    int b  = blockIdx.x / (PIXC / 64);
    int pt = blockIdx.x % (PIXC / 64);
    int pbase = pt * 64;
#pragma unroll
    for (int j = 0; j < 8; ++j) {                      // async DMA: whole cos table
        int l = t + j * 256;
        async_ld16(cosl + l * 8, cosA + l * 8);
    }
    const f16* Xb = X + (size_t)b * RES * PIXC;
    int lane = t & 31, wv = t >> 5;
    v8f acc[4];
#pragma unroll
    for (int s = 0; s < 4; ++s) acc[s] = vzero8();
    int cc = t & 63, rb8 = (t >> 6) * 8;               // transpose-staging ownership
    for (int k0 = 0; k0 < RES; k0 += 32) {
        __syncthreads();
        v8h tmp;                                       // gather a k-column, store one b128
#pragma unroll
        for (int j = 0; j < 8; ++j)
            tmp[j] = Xb[(size_t)(k0 + rb8 + j) * PIXC + pbase + cc];
        *(v8h*)&xtT[cc][rb8] = tmp;
        if (k0 == 0) async_wait_all();                 // cos table resident before first use
        __syncthreads();
        v16h a = load_a_frag(cosl + (size_t)(wv * 16) * RES, RES, k0, lane);
#pragma unroll
        for (int s = 0; s < 4; ++s) {
            v16h bf = load_b_frag_nk(&xtT[0][0], 32, s * 16, 0, lane);
            acc[s] = wmma16(a, bf, acc[s]);
        }
    }
    __syncthreads();                                   // cos table dead -> out staging
    f16* outs = cosl;                                  // [128 n][64 p]
    int col = lane & 15, half = lane >> 4;
#pragma unroll
    for (int s = 0; s < 4; ++s)
#pragma unroll
        for (int r = 0; r < 8; ++r)
            outs[(wv * 16 + r + 8 * half) * 64 + s * 16 + col] = (f16)acc[s][r];
    __syncthreads();
    f16* Yb = Y + (size_t)b * RES * PIXC;
#pragma unroll
    for (int j = 0; j < 4; ++j) {                      // 1024 chunks, contiguous per row
        int l = t + j * 256;
        int n = l >> 3, off = (l & 7) * 8;
        async_st16(Yb + (size_t)n * PIXC + pbase + off, outs + n * 64 + off);
    }
}

// ------- K6: per (b,n): T = cosm @ Xf1[b,n]; T *= wexp[n,m]^k[b,c]; U = cosm^T @ T -------

__global__ void k_dct_w_decay(const f16* __restrict__ cosm, const f16* __restrict__ cosmT,
                              const f16* __restrict__ Xf1, const float* __restrict__ lwexp,
                              const float* __restrict__ kk, f16* __restrict__ Xs2) {
    // grid = BATCH*RES; 256 threads = 8 waves; both GEMMs are 128 x 96 x 128
    __shared__ __align__(16) f16 At[RES][32];          // 8 KB  A k-slab [m][k]
    __shared__ __align__(16) f16 BtT[HID][32];         // 6 KB  B^T slab [c][k]
    __shared__ __align__(16) f16 TlT[HID][RES];        // 24 KB T^T [c][m]; reused out [w][c]
    int blk = blockIdx.x;
    int b = blk / RES, n = blk % RES;
    int t = threadIdx.x, lane = t & 31, wv = t >> 5;
    const f16* Xbn = Xf1 + (size_t)(b * RES + n) * RES * HID;
    int col16 = lane & 15, half = lane >> 4;

    v8f acc[6];
#pragma unroll
    for (int s = 0; s < 6; ++s) acc[s] = vzero8();
    for (int k0 = 0; k0 < RES; k0 += 32) {
        __syncthreads();
#pragma unroll
        for (int j = 0; j < 2; ++j) {                  // async DMA A slab: 512 chunks
            int l = t + j * 256;
            int m = l >> 2, c8 = (l & 3) * 8;
            async_ld16(&At[m][c8], cosm + m * RES + k0 + c8);
        }
        for (int i = t; i < HID * 4; i += 256) {       // transpose-stage B slab
            int c = i % HID, rb = (i / HID) * 8;
            v8h tmp;
#pragma unroll
            for (int j = 0; j < 8; ++j)
                tmp[j] = Xbn[(size_t)(k0 + rb + j) * HID + c];
            *(v8h*)&BtT[c][rb] = tmp;
        }
        async_wait_all();
        __syncthreads();
        v16h a = load_a_frag(&At[wv * 16][0], 32, 0, lane);
#pragma unroll
        for (int s = 0; s < 6; ++s) {
            v16h bf = load_b_frag_nk(&BtT[0][0], 32, s * 16, 0, lane);
            acc[s] = wmma16(a, bf, acc[s]);
        }
    }
    __syncthreads();
    // decay, park T^T in LDS (vectorized: fixed c per lane, 8 contiguous m per store)
    int mb = wv * 16 + 8 * half;
#pragma unroll
    for (int s = 0; s < 6; ++s) {
        int c = s * 16 + col16;
        float kc = kk[b * DIM + c];
        v8h tv;
#pragma unroll
        for (int r = 0; r < 8; ++r)
            tv[r] = (f16)(acc[s][r] * exp2f(lwexp[n * RES + mb + r] * kc));
        *(v8h*)&TlT[c][mb] = tv;
    }
    __syncthreads();

    v8f acc2[6];
#pragma unroll
    for (int s = 0; s < 6; ++s) acc2[s] = vzero8();
    for (int k0 = 0; k0 < RES; k0 += 32) {
        __syncthreads();
#pragma unroll
        for (int j = 0; j < 2; ++j) {                  // async DMA A slab (cosm^T)
            int l = t + j * 256;
            int m = l >> 2, c8 = (l & 3) * 8;
            async_ld16(&At[m][c8], cosmT + m * RES + k0 + c8);
        }
        async_wait_all();
        __syncthreads();
        v16h a = load_a_frag(&At[wv * 16][0], 32, 0, lane);
#pragma unroll
        for (int s = 0; s < 6; ++s) {
            v16h bf = load_b_frag_nk(&TlT[0][0], RES, s * 16, k0, lane);
            acc2[s] = wmma16(a, bf, acc2[s]);
        }
    }
    __syncthreads();                                   // T^T dead -> out staging [w][c]
    f16* outs = &TlT[0][0];
#pragma unroll
    for (int s = 0; s < 6; ++s)
#pragma unroll
        for (int r = 0; r < 8; ++r)
            outs[(wv * 16 + r + 8 * half) * HID + s * 16 + col16] = (f16)acc2[s][r];
    __syncthreads();
    f16* Yb = Xs2 + (size_t)(b * RES + n) * RES * HID;
#pragma unroll
    for (int j = 0; j < 6; ++j) {                      // 1536 chunks, fully contiguous
        int l = t + j * 256;
        async_st16(Yb + l * 8, outs + l * 8);
    }
}

// ------- K8: LN -> *silu(z) -> @ out_w^T + out_b -> NCHW f32 (LDS transpose) -------

__global__ void k_out(const f16* __restrict__ xs, const f16* __restrict__ z,
                      const f16* __restrict__ outw, const float* __restrict__ ln_g,
                      const float* __restrict__ ln_b, const float* __restrict__ out_b,
                      float* __restrict__ out) {
    // grid = BATCH * (RES*RES/64); 256 threads = 8 waves; 64 pixels x 96 outs
    __shared__ __align__(16) f16 Atile[64][HID];       // 12 KB (xs, then gated A' in place)
    __shared__ __align__(16) f16 Bt[DIM][HID];         // 18 KB out_w as [o][c]
    __shared__ float gl[HID], bl[HID], obl[DIM];
    __shared__ float mu[64], rs[64];
    __shared__ __align__(16) float Ot[DIM][64];        // 24 KB, [o][pixel] for NCHW stores
    int t = threadIdx.x;
    int b  = blockIdx.x / (RES * RES / 64);
    int pt = blockIdx.x % (RES * RES / 64);
    size_t pixbase = (size_t)b * RES * RES + (size_t)pt * 64;
    f16* Af = &Atile[0][0];
    f16* Bf = &Bt[0][0];
#pragma unroll
    for (int j = 0; j < 3; ++j) {                      // 768 chunks of xs
        int l = t + j * 256;
        async_ld16(Af + l * 8, xs + pixbase * HID + l * 8);
    }
    for (int l = t; l < (DIM * HID) / 8; l += 256)     // 1152 chunks of out_w
        async_ld16(Bf + l * 8, outw + l * 8);
    if (t < HID) { gl[t] = ln_g[t]; bl[t] = ln_b[t]; }
    if (t < DIM) obl[t] = out_b[t];
    async_wait_all();
    __syncthreads();
    if (t < 64) {
        float s = 0.0f, s2 = 0.0f;
#pragma unroll 4
        for (int c = 0; c < HID; ++c) {
            float v = (float)Atile[t][c];
            s += v; s2 += v * v;
        }
        float m = s / HID;
        mu[t] = m;
        rs[t] = rsqrtf(s2 / HID - m * m + 1e-5f);
    }
    __syncthreads();
    for (int i = t; i < 64 * HID; i += 256) {
        int p = i / HID, c = i % HID;
        float v  = ((float)Atile[p][c] - mu[p]) * rs[p] * gl[c] + bl[c];
        float zz = (float)z[pixbase * HID + i];
        float sz = zz / (1.0f + expf(-zz));            // silu
        Atile[p][c] = (f16)(v * sz);
    }
    __syncthreads();
    int lane = t & 31, wv = t >> 5;
    int col = lane & 15, half = lane >> 4;
#pragma unroll
    for (int i = 0; i < 3; ++i) {                      // 24 tiles / 8 waves
        int id = wv * 3 + i;
        int ms = id / 6, nt = id % 6;
        v8f acc = vzero8();
#pragma unroll
        for (int k0 = 0; k0 < HID; k0 += 32) {
            v16h a  = load_a_frag(&Atile[ms * 16][0], HID, k0, lane);
            v16h bf = load_b_frag_nk(Bf, HID, nt * 16, k0, lane);
            acc = wmma16(a, bf, acc);
        }
        int o = nt * 16 + col;
#pragma unroll
        for (int r = 0; r < 8; ++r)
            Ot[o][ms * 16 + r + 8 * half] = acc[r] + obl[o];
    }
    __syncthreads();
    float* ob = out + (size_t)b * DIM * RES * RES + (size_t)pt * 64;
#pragma unroll
    for (int j = 0; j < 6; ++j) {                      // 1536 x 16B chunks of f32 output
        int l = t + j * 256;
        int o = l >> 4, off = (l & 15) * 4;
        async_st16(ob + (size_t)o * RES * RES + off, &Ot[o][off]);
    }
}

// ---------------- host ----------------

extern "C" void kernel_launch(void* const* d_in, const int* in_sizes, int n_in,
                              void* d_out, int out_size, void* d_ws, size_t ws_size,
                              hipStream_t stream) {
    (void)in_sizes; (void)n_in; (void)out_size; (void)ws_size;
    const float* x      = (const float*)d_in[0];
    const float* fe     = (const float*)d_in[1];
    const float* dwc_w  = (const float*)d_in[2];
    const float* dwc_b  = (const float*)d_in[3];
    const float* lin_w  = (const float*)d_in[4];
    const float* lin_b  = (const float*)d_in[5];
    const float* tok_w  = (const float*)d_in[6];
    const float* tok_b  = (const float*)d_in[7];
    const float* ln_g   = (const float*)d_in[8];
    const float* ln_b   = (const float*)d_in[9];
    const float* out_w  = (const float*)d_in[10];
    const float* out_b  = (const float*)d_in[11];
    float* out = (float*)d_out;

    char* ws = (char*)d_ws;
    auto carve = [&](size_t bytes) {
        char* p = ws;
        ws += (bytes + 255) & ~(size_t)255;
        return p;
    };
    f16*   cosn_h  = (f16*)  carve((size_t)RES * RES * sizeof(f16));
    f16*   cosnT_h = (f16*)  carve((size_t)RES * RES * sizeof(f16));
    float* lwexp   = (float*)carve((size_t)RES * RES * sizeof(float));
    f16*   linw_h  = (f16*)  carve((size_t)2 * HID * DIM * sizeof(f16));
    f16*   outw_h  = (f16*)  carve((size_t)DIM * HID * sizeof(f16));
    float* kk      = (float*)carve((size_t)BATCH * DIM * sizeof(float));
    size_t big = (size_t)BATCH * RES * RES * HID * sizeof(f16);   // ~50.3 MB
    f16* buf0 = (f16*)carve(big);
    f16* buf1 = (f16*)carve(big);
    f16* buf2 = (f16*)carve(big);

    k_precompute<<<72, 256, 0, stream>>>(lin_w, out_w, cosn_h, cosnT_h, lwexp, linw_h, outw_h);
    k_tok<<<(BATCH * DIM + 255) / 256, 256, 0, stream>>>(fe, tok_w, tok_b, kk);
    k_dwconv<<<BATCH * RES, 256, 0, stream>>>(x, dwc_w, dwc_b, buf0);
    k_lin<<<(BATCH * RES * RES) / 64, 256, 0, stream>>>(buf0, linw_h, lin_b, buf1, buf2);
    k_dct_h<<<BATCH * (PIXC / 64), 256, 0, stream>>>(cosn_h, buf1, buf0);            // forward H
    k_dct_w_decay<<<BATCH * RES, 256, 0, stream>>>(cosn_h, cosnT_h, buf0, lwexp, kk, buf1);
    k_dct_h<<<BATCH * (PIXC / 64), 256, 0, stream>>>(cosnT_h, buf1, buf0);           // inverse H
    k_out<<<BATCH * (RES * RES / 64), 256, 0, stream>>>(buf0, buf2, outw_h, ln_g, ln_b, out_b, out);
}
